// SE3TransformerLight_4114578670356
// MI455X (gfx1250) — matrix operand order repaired and checked by reading, not compiled, not run on hardware
//
#include <hip/hip_runtime.h>
#include <hip/hip_bf16.h>
#include <math.h>

// ---------------------------------------------------------------------------
// SE(3)-Transformer-light forward for MI455X (gfx1250, wave32).
//
//   cg_init  (device-computed real CG tensors, 1000 floats)
//   geom     (edge vec -> sh[16], gaussian basis[10]*inrad, dist)
//   embed    (feat @ emb -> fs l=0 block, l=1,2 zero)
//   prep_mlp (f32 MLP weights -> fragment-ready padded f16 B-operands)
//   for layer in 0..4:
//     q_linear
//     for each 4000-node / 64000-edge chunk (L2-resident working set):
//       mlp_wmma (k weights)   -- v_wmma_f32_16x16x32_f16 x8 per 16-edge tile
//       mlp_wmma (v weights)
//       tp_edge  (CG TP via shared T[m][q]; k->logit and v in one pass;
//                 CG table staged with global_load_async_to_lds_b128)
//       attn     (segment softmax over 16 contiguous edges, aggregate)
//     node_update (Wsi + gate(Wl1) + Wl2 + skip + equivariant LN)
// ---------------------------------------------------------------------------

typedef __attribute__((ext_vector_type(16))) _Float16 v16h;
typedef __attribute__((ext_vector_type(8)))  float    v8f;

#define NPATHS 18
static constexpr int P_L1[NPATHS]    = {0,0,0,1,1,1,1,1,1,1,2,2,2,2,2,2,2,2};
static constexpr int P_L2[NPATHS]    = {0,1,2,0,1,1,1,2,2,3,0,1,1,2,2,2,3,3};
static constexpr int P_L3[NPATHS]    = {0,1,2,1,0,1,2,1,2,2,2,1,2,0,1,2,1,2};
static constexpr int P_CGOFF[NPATHS] = {0,1,10,35,44,53,80,125,170,245,350,375,420,495,520,595,720,825};
static constexpr int OFFL[4]  = {0,5,20,45};   // feature concat offsets per l
static constexpr int SHOFF[4] = {0,1,4,9};     // sh offsets per l
#define CGTOTAL 1000

__device__ __forceinline__ float fast_rcp(float x)  { return __builtin_amdgcn_rcpf(x); }
__device__ __forceinline__ float fast_rsq(float x)  { return __builtin_amdgcn_rsqf(x); }
__device__ __forceinline__ float sigm(float x)      { return fast_rcp(1.0f + __expf(-x)); }

// ======================= CG tensors (computed on device) ====================
__device__ double dfact(int n) { double r = 1.0; for (int i = 2; i <= n; ++i) r *= (double)i; return r; }

__device__ double cg_coef(int l1, int m1, int l2, int m2, int l3, int m3) {
    if (m1 + m2 != m3) return 0.0;
    double pref = sqrt((double)(2 * l3 + 1) * dfact(l3 + l1 - l2) * dfact(l3 - l1 + l2) *
                       dfact(l1 + l2 - l3) / dfact(l1 + l2 + l3 + 1));
    pref *= sqrt(dfact(l3 + m3) * dfact(l3 - m3) * dfact(l1 - m1) * dfact(l1 + m1) *
                 dfact(l2 - m2) * dfact(l2 + m2));
    int kmax = l1 + l2 - l3;
    if (l1 - m1 < kmax) kmax = l1 - m1;
    if (l2 + m2 < kmax) kmax = l2 + m2;
    double s = 0.0;
    for (int k = 0; k <= kmax; ++k) {
        if (l3 - l2 + m1 + k < 0 || l3 - l1 - m2 + k < 0) continue;
        double den = dfact(k) * dfact(l1 + l2 - l3 - k) * dfact(l1 - m1 - k) *
                     dfact(l2 + m2 - k) * dfact(l3 - l2 + m1 + k) * dfact(l3 - l1 - m2 + k);
        s += ((k & 1) ? -1.0 : 1.0) / den;
    }
    return pref * s;
}

__device__ void buildU(int l, double Ur[7][7], double Ui[7][7]) {
    for (int a = 0; a < 7; ++a)
        for (int b = 0; b < 7; ++b) { Ur[a][b] = 0.0; Ui[a][b] = 0.0; }
    Ur[l][l] = 1.0;
    double s = 1.0 / sqrt(2.0);
    for (int m = 1; m <= l; ++m) {
        double sg = (m & 1) ? -1.0 : 1.0;
        Ur[l + m][l + m] = sg * s;      // (-1)^m * s
        Ur[l + m][l - m] = s;           // s
        Ui[l - m][l + m] = -sg * s;     // -1j*(-1)^m*s
        Ui[l - m][l - m] = s;           // 1j*s
    }
}

__global__ void cg_init_kernel(float* __restrict__ cgbuf) {
    int p = threadIdx.x;
    if (p >= NPATHS) return;
    int l1 = P_L1[p], l2 = P_L2[p], l3 = P_L3[p];
    int d1 = 2 * l1 + 1, d2 = 2 * l2 + 1, d3 = 2 * l3 + 1;

    double Cc[7][7][7];
    for (int a = 0; a < d1; ++a) for (int b = 0; b < d2; ++b) for (int c = 0; c < d3; ++c) Cc[a][b][c] = 0.0;
    for (int m1 = -l1; m1 <= l1; ++m1)
        for (int m2 = -l2; m2 <= l2; ++m2) {
            int m3 = m1 + m2;
            if (m3 >= -l3 && m3 <= l3)
                Cc[m1 + l1][m2 + l2][m3 + l3] = cg_coef(l1, m1, l2, m2, l3, m3);
        }
    double U1r[7][7], U1i[7][7], U2r[7][7], U2i[7][7], U3r[7][7], U3i[7][7];
    buildU(l1, U1r, U1i); buildU(l2, U2r, U2i); buildU(l3, U3r, U3i);

    double Cr[7][7][7], Ci[7][7][7];
    for (int a = 0; a < d1; ++a)
        for (int b = 0; b < d2; ++b)
            for (int c = 0; c < d3; ++c) {
                double sr = 0.0, si = 0.0;
                for (int m = 0; m < d1; ++m)
                    for (int n = 0; n < d2; ++n)
                        for (int q = 0; q < d3; ++q) {
                            double cc = Cc[m][n][q];
                            if (cc == 0.0) continue;
                            double zr = U1r[a][m] * U2r[b][n] - U1i[a][m] * U2i[b][n];
                            double zi = U1r[a][m] * U2i[b][n] + U1i[a][m] * U2r[b][n];
                            double wr = zr * U3r[c][q] + zi * U3i[c][q];   // * conj(U3)
                            double wi = zi * U3r[c][q] - zr * U3i[c][q];
                            sr += wr * cc; si += wi * cc;
                        }
                Cr[a][b][c] = sr; Ci[a][b][c] = si;
            }
    double mr = 0.0, mi = 0.0;
    for (int a = 0; a < d1; ++a) for (int b = 0; b < d2; ++b) for (int c = 0; c < d3; ++c) {
        double xr = fabs(Cr[a][b][c]); if (xr > mr) mr = xr;
        double xi = fabs(Ci[a][b][c]); if (xi > mi) mi = xi;
    }
    bool useIm = (mi > mr);
    double nrm = 0.0;
    for (int a = 0; a < d1; ++a) for (int b = 0; b < d2; ++b) for (int c = 0; c < d3; ++c) {
        double v = useIm ? Ci[a][b][c] : Cr[a][b][c];
        nrm += v * v;
    }
    nrm = sqrt(nrm);
    double inv = (nrm > 0.0) ? 1.0 / nrm : 0.0;
    float* dst = cgbuf + P_CGOFF[p];
    for (int a = 0; a < d1; ++a) for (int b = 0; b < d2; ++b) for (int c = 0; c < d3; ++c)
        dst[(a * d2 + b) * d3 + c] = (float)((useIm ? Ci[a][b][c] : Cr[a][b][c]) * inv);
}

// ============================ edge geometry ================================
__global__ void geom_kernel(const float* __restrict__ pos,
                            const int* __restrict__ esrc, const int* __restrict__ edst,
                            float* __restrict__ shout, float* __restrict__ basis,
                            float* __restrict__ dist, int E) {
    int e = blockIdx.x * blockDim.x + threadIdx.x;
    if (e >= E) return;
    int s = esrc[e], d = edst[e];
    float vx = pos[3 * s + 0] - pos[3 * d + 0];
    float vy = pos[3 * s + 1] - pos[3 * d + 1];
    float vz = pos[3 * s + 2] - pos[3 * d + 2];
    float dd = sqrtf(vx * vx + vy * vy + vz * vz);
    float inv = fast_rcp(dd + 1e-9f);
    float x = vx * inv, y = vy * inv, z = vz * inv;

    const float s3 = 1.7320508075688772f, s6 = 2.449489742783178f;
    const float s10 = 3.1622776601683795f, s15 = 3.872983346207417f;
    float* sh = shout + (size_t)e * 16;
    sh[0] = 1.0f;
    sh[1] = y; sh[2] = z; sh[3] = x;
    sh[4] = s3 * x * y;
    sh[5] = s3 * y * z;
    sh[6] = 0.5f * (3.0f * z * z - 1.0f);
    sh[7] = s3 * x * z;
    sh[8] = 0.5f * s3 * (x * x - y * y);
    sh[9]  = 0.25f * s10 * y * (3.0f * x * x - y * y);
    sh[10] = s15 * x * y * z;
    sh[11] = 0.25f * s6 * y * (5.0f * z * z - 1.0f);
    sh[12] = 0.5f * (5.0f * z * z * z - 3.0f * z);
    sh[13] = 0.25f * s6 * x * (5.0f * z * z - 1.0f);
    sh[14] = 0.5f * s15 * z * (x * x - y * y);
    sh[15] = 0.25f * s10 * x * (x * x - 3.0f * y * y);

    dist[e] = dd;
    float msk = (dd < 2.0f) ? 1.0f : 0.0f;
    float* bp = basis + (size_t)e * 10;
    #pragma unroll
    for (int b = 0; b < 10; ++b) {
        float c = (2.0f / 9.0f) * (float)b;
        float t = (dd - c) * (9.0f / 2.0f);
        bp[b] = __expf(-t * t) * msk;
    }
}

// ============================== embedding ==================================
__global__ void embed_kernel(const float* __restrict__ feat, const float* __restrict__ emb,
                             float* __restrict__ fs, int N) {
    int n = blockIdx.x * blockDim.x + threadIdx.x;
    if (n >= N) return;
    const float* x = feat + (size_t)n * 3;
    float* o = fs + (size_t)n * 45;
    #pragma unroll
    for (int v = 0; v < 5; ++v)
        o[v] = x[0] * emb[0 * 5 + v] + x[1] * emb[1 * 5 + v] + x[2] * emb[2 * 5 + v];
    for (int i = 5; i < 45; ++i) o[i] = 0.0f;
}

// ============ MLP weight prep: f32 -> fragment-ready padded f16 ============
// B-fragment layout (CDNA5 ISA 7.12.2, 16-bit B 32x16): lane = N%16 + 16*?,
// per-lane 16 halves holding K = 8*(lane>>4) + e (e<8), 16 + 8*(lane>>4) + (e-8) (e>=8).
// w1h: [j<2][lane<32][e<16]  from W1 (10x32), K-padded to 32.
// w2h: [j<6][lane<32][e<16]  from W2 (32x90), N-padded to 96.
__global__ void prep_mlp_kernel(const float* __restrict__ W1, const float* __restrict__ W2,
                                _Float16* __restrict__ w1h, _Float16* __restrict__ w2h) {
    int t = blockIdx.x * blockDim.x + threadIdx.x;
    if (t < 1024) {
        int e = t & 15, lane = (t >> 4) & 31, j = t >> 9;
        int K = (e < 8) ? ((lane >> 4) * 8 + e) : (16 + (lane >> 4) * 8 + (e - 8));
        int Nc = (lane & 15) + 16 * j;
        w1h[t] = (_Float16)((K < 10) ? W1[K * 32 + Nc] : 0.0f);
    }
    if (t < 3072) {
        int e = t & 15, lane = (t >> 4) & 31, j = t >> 9;
        int K = (e < 8) ? ((lane >> 4) * 8 + e) : (16 + (lane >> 4) * 8 + (e - 8));
        int Nc = (lane & 15) + 16 * j;
        w2h[t] = (_Float16)((Nc < 90) ? W2[K * 90 + Nc] : 0.0f);
    }
}

// ============================ radial MLP via WMMA ==========================
#define MLP_WAVES 8
__global__ void mlp_wmma_kernel(const float* __restrict__ basis, const float* __restrict__ b1,
                                const _Float16* __restrict__ w1h, const _Float16* __restrict__ w2h,
                                float* __restrict__ wout, int chunkEdgeBase, int numTiles) {
    __shared__ _Float16 hTile[MLP_WAVES][16][34];   // +2 halves pad per row

    int lane = threadIdx.x & 31;
    int wv   = threadIdx.x >> 5;
    int tile = blockIdx.x * MLP_WAVES + wv;
    if (tile >= numTiles) tile = numTiles - 1;      // wave-uniform clamp (duplicate, same values)
    int row0 = tile * 16;                           // chunk-local edge row

    int m     = lane & 15;
    int kbase = (lane >> 4) * 8;                    // A-fragment K base per lane-half

    // --- A fragment: contiguous dword loads of this lane's basis row slice ---
    const float* p = basis + (size_t)(chunkEdgeBase + row0 + m) * 10 + kbase;
    float vals[8];
    #pragma unroll
    for (int i = 0; i < 8; ++i) vals[i] = p[i];
    v16h a;
    #pragma unroll
    for (int e = 0; e < 16; ++e) {
        float x = (e < 8 && (kbase + e) < 10) ? vals[e] : 0.0f;   // K>=10 zero-padded
        a[e] = (_Float16)x;
    }

    const v16h* fragW1 = (const v16h*)w1h;          // [2][32]
    const v16h* fragW2 = (const v16h*)w2h;          // [6][32]

    // --- GEMM1: two N-tiles of W1 ---
    v8f c01[2];
    #pragma unroll
    for (int j = 0; j < 2; ++j) {
        v16h b = fragW1[j * 32 + lane];             // single 32B aligned load
        v8f z = {0.f, 0.f, 0.f, 0.f, 0.f, 0.f, 0.f, 0.f};
        c01[j] = __builtin_amdgcn_wmma_f32_16x16x32_f16(false, a, false, b, (short)0, z, false, false);
    }

    // --- bias + silu, re-layout through LDS (C layout -> A layout) ---
    #pragma unroll
    for (int j = 0; j < 2; ++j) {
        int Nc = (lane & 15) + 16 * j;
        float bias = b1[Nc];
        #pragma unroll
        for (int r = 0; r < 8; ++r) {
            float h = c01[j][r] + bias;
            h = h * sigm(h);                        // silu
            int M = r + 8 * (lane >> 4);
            hTile[wv][M][Nc] = (_Float16)h;
        }
    }
    __syncthreads();

    v16h a2;
    #pragma unroll
    for (int e = 0; e < 16; ++e) {
        int k = (e < 8) ? (kbase + e) : (16 + kbase + (e - 8));
        a2[e] = hTile[wv][m][k];
    }

    // --- GEMM2: six N-tiles of W2, scatter D (N < 90) ---
    #pragma unroll
    for (int j = 0; j < 6; ++j) {
        v16h b = fragW2[j * 32 + lane];
        v8f z = {0.f, 0.f, 0.f, 0.f, 0.f, 0.f, 0.f, 0.f};
        v8f d = __builtin_amdgcn_wmma_f32_16x16x32_f16(false, a2, false, b, (short)0, z, false, false);
        int Nc = (lane & 15) + 16 * j;
        if (Nc < 90) {
            #pragma unroll
            for (int r = 0; r < 8; ++r) {
                int M = r + 8 * (lane >> 4);
                wout[(size_t)(row0 + M) * 90 + Nc] = d[r];
            }
        }
    }
}

// ============================ per-node q linear ============================
__global__ void q_linear_kernel(const float* __restrict__ fs,
                                const float* __restrict__ W0, const float* __restrict__ W1,
                                const float* __restrict__ W2,
                                float* __restrict__ q, int N) {
    int n = blockIdx.x * blockDim.x + threadIdx.x;
    if (n >= N) return;
    const float* f = fs + (size_t)n * 45;
    float* o = q + (size_t)n * 45;
    #pragma unroll
    for (int v = 0; v < 5; ++v) {
        float s = 0.f;
        for (int u = 0; u < 5; ++u) s += f[u] * W0[u * 5 + v];
        o[v] = s;
    }
    for (int v = 0; v < 5; ++v)
        for (int mm = 0; mm < 3; ++mm) {
            float s = 0.f;
            for (int u = 0; u < 5; ++u) s += f[5 + u * 3 + mm] * W1[u * 5 + v];
            o[5 + v * 3 + mm] = s;
        }
    for (int v = 0; v < 5; ++v)
        for (int mm = 0; mm < 5; ++mm) {
            float s = 0.f;
            for (int u = 0; u < 5; ++u) s += f[20 + u * 5 + mm] * W2[u * 5 + v];
            o[20 + v * 5 + mm] = s;
        }
}

// ====================== fused tensor product per edge ======================
// Per path: T[m][q] = sum_n sh[n]*CG[m][n][q] (channel independent), then
// s[q] = sum_m f[u][m]*T[m][q] reused for BOTH k and v accumulation.
// Fully unrolled (constexpr tables) -> ds_load with immediate offsets.
__global__ void tp_edge_kernel(const float* __restrict__ fs, const float* __restrict__ sh_all,
                               const float* __restrict__ wk, const float* __restrict__ wvw,
                               const float* __restrict__ qcat, const float* __restrict__ dist,
                               const int* __restrict__ esrc, const int* __restrict__ edst,
                               const float* __restrict__ cg_global,
                               int chunkEdgeBase, int chunkEdges,
                               float* __restrict__ logit_out, float* __restrict__ v_out) {
    __shared__ float CG[CGTOTAL];
    // Stage the 4KB CG table via async DMA to LDS (ASYNCcnt path):
    // 250 lanes x 16B. Flat LDS addresses: low 32 bits are the LDS byte offset.
    if (threadIdx.x < CGTOTAL / 4) {
        unsigned ldsOff = (unsigned)(size_t)(&CG[threadIdx.x * 4]);
        const float* gp = cg_global + (size_t)threadIdx.x * 4;
        asm volatile("global_load_async_to_lds_b128 %0, %1, off"
                     :: "v"(ldsOff), "v"(gp) : "memory");
    }
    asm volatile("s_wait_asynccnt 0x0" ::: "memory");
    __syncthreads();

    int t = blockIdx.x * blockDim.x + threadIdx.x;
    if (t >= chunkEdges) return;
    int ge  = chunkEdgeBase + t;
    int src = esrc[ge];
    int dn  = edst[ge];

    __builtin_prefetch(qcat + (size_t)dn * 45, 0, 1);   // global_prefetch_b8

    float f[45], shv[16], outk[45], outv[45];
    const float* fp = fs + (size_t)src * 45;
    #pragma unroll
    for (int i = 0; i < 45; ++i) f[i] = fp[i];
    const float* sp = sh_all + (size_t)ge * 16;
    #pragma unroll
    for (int i = 0; i < 16; ++i) shv[i] = sp[i];
    #pragma unroll
    for (int i = 0; i < 45; ++i) { outk[i] = 0.f; outv[i] = 0.f; }

    const float* wkr = wk  + (size_t)t * 90;
    const float* wvr = wvw + (size_t)t * 90;

    #pragma unroll
    for (int p = 0; p < NPATHS; ++p) {
        const int l1 = P_L1[p], l2 = P_L2[p], l3 = P_L3[p];
        const int d1 = 2 * l1 + 1, d2 = 2 * l2 + 1, d3 = 2 * l3 + 1;
        const float* C  = CG + P_CGOFF[p];
        const float* sh = shv + SHOFF[l2];

        float T[25];                                 // d1*d3 <= 25
        #pragma unroll
        for (int mm = 0; mm < d1; ++mm)
            #pragma unroll
            for (int q = 0; q < d3; ++q) {
                float acc = 0.f;
                #pragma unroll
                for (int nn = 0; nn < d2; ++nn)
                    acc += sh[nn] * C[(mm * d2 + nn) * d3 + q];
                T[mm * d3 + q] = acc;
            }

        #pragma unroll
        for (int u = 0; u < 5; ++u) {
            float wku = wkr[p * 5 + u];
            float wvu = wvr[p * 5 + u];
            const float* fu = f + OFFL[l1] + u * d1;
            float s[5];
            #pragma unroll
            for (int q = 0; q < d3; ++q) s[q] = 0.f;
            #pragma unroll
            for (int mm = 0; mm < d1; ++mm) {
                float fv = fu[mm];
                #pragma unroll
                for (int q = 0; q < d3; ++q) s[q] += fv * T[mm * d3 + q];
            }
            const int ob = OFFL[l3] + u * d3;
            #pragma unroll
            for (int q = 0; q < d3; ++q) {
                outk[ob + q] += wku * s[q];
                outv[ob + q] += wvu * s[q];
            }
        }
    }

    const float* qp = qcat + (size_t)dn * 45;
    float lg = 0.f;
    #pragma unroll
    for (int i = 0; i < 45; ++i) lg += outk[i] * qp[i];
    lg *= 0.14907119849998599f;                         // 1/sqrt(45)
    logit_out[t] = (dist[ge] < 2.0f) ? lg : -1e9f;

    float* vp = v_out + (size_t)t * 45;
    #pragma unroll
    for (int i = 0; i < 45; ++i) vp[i] = outv[i];
}

// ================= segment softmax + aggregation (16 edges/node) ===========
__global__ void attn_kernel(const float* __restrict__ logit, const float* __restrict__ v,
                            float* __restrict__ agg, int chunkNodeBase, int chunkNodes) {
    int t = blockIdx.x * blockDim.x + threadIdx.x;
    if (t >= chunkNodes * 45) return;
    int nl = t / 45, dim = t % 45;
    const float* lg = logit + (size_t)nl * 16;
    float l[16];
    #pragma unroll
    for (int j = 0; j < 16; ++j) l[j] = lg[j];
    float mx = -3.4e38f;
    #pragma unroll
    for (int j = 0; j < 16; ++j) mx = fmaxf(mx, l[j]);
    float ex[16], den = 0.f;
    #pragma unroll
    for (int j = 0; j < 16; ++j) { ex[j] = __expf(l[j] - mx); den += ex[j]; }
    float inv = fast_rcp(den + 1e-9f);
    float acc = 0.f;
    #pragma unroll
    for (int j = 0; j < 16; ++j)
        acc += ex[j] * v[((size_t)nl * 16 + j) * 45 + dim];
    agg[(size_t)(chunkNodeBase + nl) * 45 + dim] = acc * inv;
}

// ============ self-interaction + gate + output linear + skip + LN ==========
__global__ void node_update_kernel(const float* __restrict__ fs, const float* __restrict__ agg,
                                   const float* __restrict__ Ws0, const float* __restrict__ Ws1,
                                   const float* __restrict__ Ws2,
                                   const float* __restrict__ Wa0, const float* __restrict__ Wa1,
                                   const float* __restrict__ Wa2,
                                   const float* __restrict__ Wb0, const float* __restrict__ Wb1,
                                   const float* __restrict__ Wb2,
                                   int skip, float* __restrict__ fout, int N) {
    int n = blockIdx.x * blockDim.x + threadIdx.x;
    if (n >= N) return;
    float f[45], out[45];
    const float* fp = fs + (size_t)n * 45;
    const float* ap = agg + (size_t)n * 45;
    #pragma unroll
    for (int i = 0; i < 45; ++i) f[i] = fp[i];

    // out = agg + fs @ Wsi (per l)
    for (int v = 0; v < 5; ++v) {
        float s = 0.f;
        for (int u = 0; u < 5; ++u) s += f[u] * Ws0[u * 5 + v];
        out[v] = ap[v] + s;
    }
    for (int v = 0; v < 5; ++v)
        for (int mm = 0; mm < 3; ++mm) {
            float s = 0.f;
            for (int u = 0; u < 5; ++u) s += f[5 + u * 3 + mm] * Ws1[u * 5 + v];
            out[5 + v * 3 + mm] = ap[5 + v * 3 + mm] + s;
        }
    for (int v = 0; v < 5; ++v)
        for (int mm = 0; mm < 5; ++mm) {
            float s = 0.f;
            for (int u = 0; u < 5; ++u) s += f[20 + u * 5 + mm] * Ws2[u * 5 + v];
            out[20 + v * 5 + mm] = ap[20 + v * 5 + mm] + s;
        }

    // h = out @ Wl1 (5 -> 10 channels), gate nonlinearity
    float h0[10], h1[30], h2[50];
    for (int v = 0; v < 10; ++v) {
        float s = 0.f;
        for (int u = 0; u < 5; ++u) s += out[u] * Wa0[u * 10 + v];
        h0[v] = s * sigm(s);                         // silu
    }
    for (int v = 0; v < 10; ++v) {
        float nn = 1e-6f;
        for (int mm = 0; mm < 3; ++mm) {
            float s = 0.f;
            for (int u = 0; u < 5; ++u) s += out[5 + u * 3 + mm] * Wa1[u * 10 + v];
            h1[v * 3 + mm] = s;
            nn += s * s;
        }
        float sc = sigm(sqrtf(nn));                  // silu(n)/n == sigmoid(n)
        for (int mm = 0; mm < 3; ++mm) h1[v * 3 + mm] *= sc;
    }
    for (int v = 0; v < 10; ++v) {
        float nn = 1e-6f;
        for (int mm = 0; mm < 5; ++mm) {
            float s = 0.f;
            for (int u = 0; u < 5; ++u) s += out[20 + u * 5 + mm] * Wa2[u * 10 + v];
            h2[v * 5 + mm] = s;
            nn += s * s;
        }
        float sc = sigm(sqrtf(nn));
        for (int mm = 0; mm < 5; ++mm) h2[v * 5 + mm] *= sc;
    }

    // y = h @ Wl2 (10 -> 5)
    float y[45];
    for (int v = 0; v < 5; ++v) {
        float s = 0.f;
        for (int u = 0; u < 10; ++u) s += h0[u] * Wb0[u * 5 + v];
        y[v] = s;
    }
    for (int v = 0; v < 5; ++v)
        for (int mm = 0; mm < 3; ++mm) {
            float s = 0.f;
            for (int u = 0; u < 10; ++u) s += h1[u * 3 + mm] * Wb1[u * 5 + v];
            y[5 + v * 3 + mm] = s;
        }
    for (int v = 0; v < 5; ++v)
        for (int mm = 0; mm < 5; ++mm) {
            float s = 0.f;
            for (int u = 0; u < 10; ++u) s += h2[u * 5 + mm] * Wb2[u * 5 + v];
            y[20 + v * 5 + mm] = s;
        }
    if (skip) for (int i = 0; i < 45; ++i) y[i] += f[i];

    // equivariant layernorm
    float mu = 0.f;
    for (int v = 0; v < 5; ++v) mu += y[v];
    mu *= 0.2f;
    for (int v = 0; v < 5; ++v) y[v] -= mu;
    float ss = 0.f;
    for (int v = 0; v < 5; ++v) ss += y[v] * y[v];
    float r = fast_rsq(ss * (1.0f / 5.0f) + 1e-6f);
    for (int v = 0; v < 5; ++v) y[v] *= r;
    ss = 0.f;
    for (int i = 5; i < 20; ++i) ss += y[i] * y[i];
    r = fast_rsq(ss * (1.0f / 15.0f) + 1e-6f);
    for (int i = 5; i < 20; ++i) y[i] *= r;
    ss = 0.f;
    for (int i = 20; i < 45; ++i) ss += y[i] * y[i];
    r = fast_rsq(ss * (1.0f / 25.0f) + 1e-6f);
    for (int i = 20; i < 45; ++i) y[i] *= r;

    float* op = fout + (size_t)n * 45;
    #pragma unroll
    for (int i = 0; i < 45; ++i) op[i] = y[i];
}

// =============================== host driver ===============================
static inline int ceil_div(int a, int b) { return (a + b - 1) / b; }

extern "C" void kernel_launch(void* const* d_in, const int* in_sizes, int n_in,
                              void* d_out, int out_size, void* d_ws, size_t ws_size,
                              hipStream_t stream) {
    (void)out_size; (void)ws_size;
    // setup_inputs order: feat, pos, params(pytree sorted-key flatten), edge_src, edge_dst
    const float* feat = (const float*)d_in[0];
    const float* pos  = (const float*)d_in[1];
    const int* esrc   = (const int*)d_in[n_in - 2];
    const int* edst   = (const int*)d_in[n_in - 1];
    const int N = in_sizes[0] / 3;
    const int E = in_sizes[n_in - 1];

    // params: {'emb': (3,5), 'layers': [5 x {'Wl1'x3,'Wl2'x3,'Wq'x3,'Wsi'x3,
    //          'mlp_k':{W1,W2,b1}, 'mlp_v':{W1,W2,b1}}]} (sorted dict keys)
    const float* emb = (const float*)d_in[2];
    auto LP = [&](int L, int j) -> const float* { return (const float*)d_in[3 + L * 18 + j]; };
    // j: 0-2 Wl1[l], 3-5 Wl2[l], 6-8 Wq[l], 9-11 Wsi[l], 12/13/14 k.{W1,W2,b1}, 15/16/17 v.{W1,W2,b1}

    // ---- workspace layout ----
    char* wsb = (char*)d_ws;
    size_t off = 0;
    auto alloc = [&](size_t bytes) -> void* {
        void* p = wsb + off;
        off = (off + bytes + 255) & ~(size_t)255;
        return p;
    };
    float* cg    = (float*)alloc(CGTOTAL * sizeof(float));
    float* sh    = (float*)alloc((size_t)E * 16 * sizeof(float));
    float* basis = (float*)alloc((size_t)E * 10 * sizeof(float) + 64);  // tail pad for 8-wide reads
    float* dist  = (float*)alloc((size_t)E * sizeof(float));
    float* fsA   = (float*)alloc((size_t)N * 45 * sizeof(float));
    float* fsB   = (float*)alloc((size_t)N * 45 * sizeof(float));
    float* qcat  = (float*)alloc((size_t)N * 45 * sizeof(float));
    float* agg   = (float*)alloc((size_t)N * 45 * sizeof(float));
    const int CN = 4000;                 // chunk nodes -> 64000 edges, L2-resident
    const int DEG = E / N;               // 16
    const int CEmax = CN * DEG;
    float* wk  = (float*)alloc((size_t)CEmax * 90 * sizeof(float));
    float* wv  = (float*)alloc((size_t)CEmax * 90 * sizeof(float));
    float* vch = (float*)alloc((size_t)CEmax * 45 * sizeof(float));
    float* lg  = (float*)alloc((size_t)CEmax * sizeof(float));
    _Float16* w1h = (_Float16*)alloc((size_t)5 * 2 * 1024 * sizeof(_Float16));
    _Float16* w2h = (_Float16*)alloc((size_t)5 * 2 * 3072 * sizeof(_Float16));

    const int BT = 256;

    // ---- once-per-call init ----
    cg_init_kernel<<<1, 32, 0, stream>>>(cg);
    geom_kernel<<<ceil_div(E, BT), BT, 0, stream>>>(pos, esrc, edst, sh, basis, dist, E);
    embed_kernel<<<ceil_div(N, BT), BT, 0, stream>>>(feat, emb, fsA, N);
    for (int L = 0; L < 5; ++L)
        for (int s = 0; s < 2; ++s)
            prep_mlp_kernel<<<ceil_div(3072, BT), BT, 0, stream>>>(
                LP(L, 12 + 3 * s), LP(L, 13 + 3 * s),
                w1h + (size_t)(L * 2 + s) * 1024, w2h + (size_t)(L * 2 + s) * 3072);

    // ---- layers ----
    float* cur = fsA;
    float* nxt = fsB;
    for (int L = 0; L < 5; ++L) {
        q_linear_kernel<<<ceil_div(N, BT), BT, 0, stream>>>(cur, LP(L, 6), LP(L, 7), LP(L, 8), qcat, N);

        for (int base = 0; base < N; base += CN) {
            int cn = (N - base < CN) ? (N - base) : CN;
            int ce = cn * DEG;
            int eb = base * DEG;
            int tiles = ce / 16;
            int mblocks = ceil_div(tiles, MLP_WAVES);
            mlp_wmma_kernel<<<mblocks, 32 * MLP_WAVES, 0, stream>>>(
                basis, LP(L, 14), w1h + (size_t)(L * 2 + 0) * 1024, w2h + (size_t)(L * 2 + 0) * 3072,
                wk, eb, tiles);
            mlp_wmma_kernel<<<mblocks, 32 * MLP_WAVES, 0, stream>>>(
                basis, LP(L, 17), w1h + (size_t)(L * 2 + 1) * 1024, w2h + (size_t)(L * 2 + 1) * 3072,
                wv, eb, tiles);
            tp_edge_kernel<<<ceil_div(ce, BT), BT, 0, stream>>>(
                cur, sh, wk, wv, qcat, dist, esrc, edst, cg, eb, ce, lg, vch);
            attn_kernel<<<ceil_div(cn * 45, BT), BT, 0, stream>>>(lg, vch, agg, base, cn);
        }

        float* outp = (L == 4) ? (float*)d_out : nxt;
        node_update_kernel<<<ceil_div(N, BT), BT, 0, stream>>>(
            cur, agg, LP(L, 9), LP(L, 10), LP(L, 11),
            LP(L, 0), LP(L, 1), LP(L, 2),
            LP(L, 3), LP(L, 4), LP(L, 5),
            (L > 0) ? 1 : 0, outp, N);
        if (L < 4) { float* t = cur; cur = nxt; nxt = t; }
    }
}
